// RecipientFinder_30425548325373
// MI455X (gfx1250) — matrix operand order repaired and checked by reading, not compiled, not run on hardware
//
#include <hip/hip_runtime.h>
#include <hip/hip_bf16.h>

typedef __attribute__((ext_vector_type(16))) __bf16 v16bf;
typedef __attribute__((ext_vector_type(8)))  __bf16 v8bf;
typedef __attribute__((ext_vector_type(8)))  float  v8f;

#define NNODES 8191
#define LEAF0  4095      // first leaf index; internal nodes are 0..4094
#define DDIM   1600
#define KA     1632      // D + IDD
#define FD     3232      // 2*D + IDD
#define LDT    40        // LDS tile stride (bf16 elems), 80B -> 16B aligned rows
#define ABUF   (256 * LDT)   // elems per A buffer (256 M-rows x 32 K)
#define BBUF   (64 * LDT)    // elems per B buffer (64 N-rows x 32 K)

__device__ __forceinline__ float sigf(float x) { return 1.f / (1.f + __expf(-x)); }

// ---------------------------------------------------------------------------
// init: zero c (leaves must read c=0) and the fc0 pad row of feat (row 8191)
// ---------------------------------------------------------------------------
__global__ void k_init(float* __restrict__ c, __bf16* __restrict__ featPadRow) {
    int i = blockIdx.x * 256 + threadIdx.x;
    if (i < NNODES * DDIM) c[i] = 0.f;
    if (i < FD) featPadRow[i] = (__bf16)0.f;
}

// ---------------------------------------------------------------------------
// f32 -> bf16 weight conversion
// ---------------------------------------------------------------------------
__global__ void k_cvt(const float* __restrict__ src, __bf16* __restrict__ dst, int n) {
    int i = blockIdx.x * 256 + threadIdx.x;
    if (i < n) dst[i] = (__bf16)src[i];
}

// slice fc0_w [64,9696] columns [colOff, colOff+FD) -> bf16 [64,FD]
__global__ void k_cvt_fc0(const float* __restrict__ w, __bf16* __restrict__ dst, int colOff) {
    int i = blockIdx.x * 256 + threadIdx.x;
    if (i < 64 * FD) {
        int r = i / FD, cc = i - r * FD;
        dst[i] = (__bf16)w[r * (3 * FD) + colOff + cc];
    }
}

// internal_node_data -> feat[:, 3200:3232]
__global__ void k_cvt_idd(const float* __restrict__ idd, __bf16* __restrict__ feat) {
    int i = blockIdx.x * 256 + threadIdx.x;
    if (i < NNODES * 32) {
        int n = i >> 5, j = i & 31;
        feat[(size_t)n * FD + 3200 + j] = (__bf16)idd[i];
    }
}

// ---------------------------------------------------------------------------
// conv front-end: x_int -> feat[:,0:1600] (all nodes); x_lstm -> feat[:,1600:3200] (leaves)
// out[n, co*200+p] = relu(b[co] + sum_{k,a} x[n,25p+5k+a] * w[co*25 + a*5 + k])
// ---------------------------------------------------------------------------
__global__ void k_conv(const float* __restrict__ x,
                       const float* __restrict__ cw,  const float* __restrict__ cb,
                       const float* __restrict__ clw, const float* __restrict__ clb,
                       __bf16* __restrict__ feat) {
    int idx = blockIdx.x * 256 + threadIdx.x;
    if (idx >= NNODES * 200) return;
    int node = idx / 200, p = idx - node * 200;
    const float* xs = x + (size_t)node * 5000 + p * 25;
    float v[25];
#pragma unroll
    for (int j = 0; j < 25; ++j) v[j] = xs[j];
    __bf16* frow = feat + (size_t)node * FD;
#pragma unroll
    for (int co = 0; co < 8; ++co) {
        float s = cb[co];
#pragma unroll
        for (int k = 0; k < 5; ++k)
#pragma unroll
            for (int a = 0; a < 5; ++a)
                s += v[5 * k + a] * cw[co * 25 + a * 5 + k];
        frow[co * 200 + p] = (__bf16)fmaxf(s, 0.f);
    }
    if (node >= LEAF0) {
#pragma unroll
        for (int co = 0; co < 8; ++co) {
            float s = clb[co];
#pragma unroll
            for (int k = 0; k < 5; ++k)
#pragma unroll
                for (int a = 0; a < 5; ++a)
                    s += v[5 * k + a] * clw[co * 25 + a * 5 + k];
            frow[DDIM + co * 200 + p] = (__bf16)fmaxf(s, 0.f);  // h for leaf
        }
    }
}

// ---------------------------------------------------------------------------
// per-level gather: build A_iou = [idd, hl+hr], A_fl = [idd, hl], A_fr = [idd, hr]
// rows [M, Mpad) zero-filled so the GEMM runs unguarded
// ---------------------------------------------------------------------------
__global__ void k_gather(const __bf16* __restrict__ feat, const float* __restrict__ idd,
                         __bf16* __restrict__ Aiou, __bf16* __restrict__ Afl,
                         __bf16* __restrict__ Afr, int base, int M, int Mpad) {
    int idx = blockIdx.x * 256 + threadIdx.x;
    if (idx >= Mpad * KA) return;
    int m = idx / KA, k = idx - m * KA;
    __bf16 vio = (__bf16)0.f, vfl = (__bf16)0.f, vfr = (__bf16)0.f;
    if (m < M) {
        int node = base + m;
        if (k < 32) {
            __bf16 v = (__bf16)idd[node * 32 + k];
            vio = v; vfl = v; vfr = v;
        } else {
            int d = k - 32;
            float hl = (float)feat[(size_t)(2 * node + 1) * FD + DDIM + d];
            float hr = (float)feat[(size_t)(2 * node + 2) * FD + DDIM + d];
            vio = (__bf16)(hl + hr); vfl = (__bf16)hl; vfr = (__bf16)hr;
        }
    }
    Aiou[idx] = vio; Afl[idx] = vfl; Afr[idx] = vfr;
}

// ---------------------------------------------------------------------------
// bf16 WMMA GEMM: C[M,Nout] = A[M,K](bf16,row-major) x W[Nout,K](bf16,row-major)^T
// block = 256 threads (8 waves), block tile 256x64, K-step 32.
// Each wave owns TWO 16-row M tiles x four 16-col N tiles -> 8 WMMAs per step
// amortizing the 4 shared B fragments (12 ds_load_b128 : 8 wmma).
// Global->LDS staging uses CDNA5 async LDS DMA (global_load_async_to_lds_b128,
// ASYNCcnt) with double-buffered LDS: the fetch of step s+1 overlaps the WMMAs
// of step s; one s_wait_asynccnt 0 + one barrier per step.
// ---------------------------------------------------------------------------
__global__ __launch_bounds__(256) void k_gemm(const __bf16* __restrict__ A,
                                              const __bf16* __restrict__ W,
                                              float* __restrict__ C, int K, int Nout) {
    __shared__ __align__(16) __bf16 lA[2 * ABUF];
    __shared__ __align__(16) __bf16 lB[2 * BBUF];
    const int tid    = threadIdx.x;
    const int wave   = tid >> 5;
    const int lane   = tid & 31;
    const int laneLo = lane & 15;
    const int laneHi = (lane >> 4) & 1;
    const long blockM = (long)blockIdx.y * 256;
    const long n0     = (long)blockIdx.x * 64;

    // per-thread staging slots: A = 1024 16B-chunks (4/thread), B = 256 (1/thread)
    const int rowA0 = tid >> 2, q = tid & 3;      // A rows 0..63 (+64*j)
    const int rowB  = tid >> 2;                   // B rows 0..63
    const __bf16* gA[4];
    unsigned dA[4];
#pragma unroll
    for (int j = 0; j < 4; ++j) {
        gA[j] = A + (blockM + rowA0 + 64 * j) * (long)K + q * 8;
        // flat LDS addresses: low 32 bits = workgroup-relative LDS byte offset
        dA[j] = (unsigned)(uintptr_t)&lA[(rowA0 + 64 * j) * LDT + q * 8];
    }
    const __bf16* gB = W + (n0 + rowB) * (long)K + q * 8;
    const unsigned dB = (unsigned)(uintptr_t)&lB[rowB * LDT + q * 8];

    auto prefetch = [&](int s) {
        const unsigned sel = (unsigned)(s & 1);
        const unsigned oA  = sel * (unsigned)(ABUF * sizeof(__bf16));
        const unsigned oB  = sel * (unsigned)(BBUF * sizeof(__bf16));
        const long koff = (long)s * 32;
#pragma unroll
        for (int j = 0; j < 4; ++j)
            asm volatile("global_load_async_to_lds_b128 %0, %1, off"
                         :: "v"(dA[j] + oA),
                            "v"((unsigned long long)(uintptr_t)(gA[j] + koff)) : "memory");
        asm volatile("global_load_async_to_lds_b128 %0, %1, off"
                     :: "v"(dB + oB),
                        "v"((unsigned long long)(uintptr_t)(gB + koff)) : "memory");
    };

    v8f acc0[4], acc1[4];
#pragma unroll
    for (int t = 0; t < 4; ++t)
#pragma unroll
        for (int i = 0; i < 8; ++i) { acc0[t][i] = 0.f; acc1[t][i] = 0.f; }

    const int S = K >> 5;  // K / 32
    prefetch(0);
    for (int s = 0; s < S; ++s) {
        asm volatile("s_wait_asynccnt 0" ::: "memory");
        __syncthreads();
        if (s + 1 < S) prefetch(s + 1);

        const __bf16* bufA = lA + (s & 1) * ABUF;
        const __bf16* bufB = lB + (s & 1) * BBUF;

        // A fragments (16x32): lane holds row (lane&15); K slots per ISA layout
        int rA0 = wave * 16 + laneLo;        // M group 0
        int rA1 = 128 + wave * 16 + laneLo;  // M group 1
        v8bf a0lo = *(const v8bf*)&bufA[rA0 * LDT + laneHi * 8];
        v8bf a0hi = *(const v8bf*)&bufA[rA0 * LDT + 16 + laneHi * 8];
        v8bf a1lo = *(const v8bf*)&bufA[rA1 * LDT + laneHi * 8];
        v8bf a1hi = *(const v8bf*)&bufA[rA1 * LDT + 16 + laneHi * 8];
        v16bf af0, af1;
#pragma unroll
        for (int i = 0; i < 8; ++i) {
            af0[i] = a0lo[i]; af0[8 + i] = a0hi[i];
            af1[i] = a1lo[i]; af1[8 + i] = a1hi[i];
        }

        // B fragments (32x16): lane holds col (lane&15); K slots per ISA layout (same as A)
        v16bf bf[4];
#pragma unroll
        for (int t = 0; t < 4; ++t) {
            int rB = t * 16 + laneLo;
            v8bf blo = *(const v8bf*)&bufB[rB * LDT + laneHi * 8];
            v8bf bhi = *(const v8bf*)&bufB[rB * LDT + 16 + laneHi * 8];
#pragma unroll
            for (int i = 0; i < 8; ++i) { bf[t][i] = blo[i]; bf[t][8 + i] = bhi[i]; }
        }

#pragma unroll
        for (int t = 0; t < 4; ++t) {
            acc0[t] = __builtin_amdgcn_wmma_f32_16x16x32_bf16(
                false, af0, false, bf[t], (short)0, acc0[t], false, false);
            acc1[t] = __builtin_amdgcn_wmma_f32_16x16x32_bf16(
                false, af1, false, bf[t], (short)0, acc1[t], false, false);
        }
        // no closing barrier needed: next overwrite of this buffer is issued only
        // after the *next* step's barrier, which follows these ds_load consumers
    }

    // C/D layout: VGPR r, lanes 0-15 -> M=r, lanes 16-31 -> M=8+r; col = lane&15
#pragma unroll
    for (int t = 0; t < 4; ++t) {
        long col = n0 + t * 16 + laneLo;
#pragma unroll
        for (int r = 0; r < 8; ++r) {
            long row0 = blockM + wave * 16 + laneHi * 8 + r;
            C[row0 * (long)Nout + col] = acc0[t][r];
            C[(row0 + 128) * (long)Nout + col] = acc1[t][r];
        }
    }
}

// ---------------------------------------------------------------------------
// LSTM pointwise: gates -> c, h (h written into feat[:,1600:3200] as bf16)
// ---------------------------------------------------------------------------
__global__ void k_lstm(const float* __restrict__ Gi, const float* __restrict__ Go,
                       const float* __restrict__ Gu, const float* __restrict__ Gfl,
                       const float* __restrict__ Gfr,
                       const float* __restrict__ bi, const float* __restrict__ bo,
                       const float* __restrict__ bu, const float* __restrict__ bff,
                       float* __restrict__ c, __bf16* __restrict__ feat,
                       int base, int M) {
    int idx = blockIdx.x * 256 + threadIdx.x;
    if (idx >= M * DDIM) return;
    int m = idx / DDIM, d = idx - m * DDIM;
    int node = base + m;
    int li = 2 * node + 1, ri = 2 * node + 2;
    size_t g = (size_t)m * DDIM + d;
    float i  = sigf(Gi[g]  + bi[d]);
    float o  = sigf(Go[g]  + bo[d]);
    float u  = tanhf(Gu[g] + bu[d]);
    float fl = sigf(Gfl[g] + bff[d]);
    float fr = sigf(Gfr[g] + bff[d]);
    float cn = i * u + fl * c[(size_t)li * DDIM + d] + fr * c[(size_t)ri * DDIM + d];
    float hn = o * tanhf(cn);
    c[(size_t)node * DDIM + d] = cn;
    feat[(size_t)node * FD + DDIM + d] = (__bf16)hn;
}

// ---------------------------------------------------------------------------
// tail: combine fc0 terms (self/parent/child-mean), relu, fc1+relu, fc2
// one 64-thread block per node
// ---------------------------------------------------------------------------
__global__ __launch_bounds__(64) void k_tail(const float* __restrict__ y0,
                                             const float* __restrict__ y1,
                                             const float* __restrict__ y2,
                                             const float* __restrict__ fc0b,
                                             const float* __restrict__ fc1w,
                                             const float* __restrict__ fc1b,
                                             const float* __restrict__ fc2w,
                                             const float* __restrict__ fc2b,
                                             float* __restrict__ out) {
    __shared__ float z0[64];
    __shared__ float red[64];
    int n = blockIdx.x, j = threadIdx.x;
    float t = y0[(size_t)n * 64 + j] + fc0b[j];
    if (n > 0) t += y1[(size_t)((n - 1) >> 1) * 64 + j];
    if (n < LEAF0)
        t += 0.5f * (y2[(size_t)(2 * n + 1) * 64 + j] + y2[(size_t)(2 * n + 2) * 64 + j]);
    z0[j] = fmaxf(t, 0.f);
    __syncthreads();
    float s = fc1b[j];
#pragma unroll 8
    for (int k = 0; k < 64; ++k) s += fc1w[j * 64 + k] * z0[k];
    red[j] = fmaxf(s, 0.f) * fc2w[j];
    __syncthreads();
    for (int off = 32; off > 0; off >>= 1) {
        if (j < off) red[j] += red[j + off];
        __syncthreads();
    }
    if (j == 0) out[n] = red[0] + fc2b[0];
}

// ---------------------------------------------------------------------------
// host driver
// ---------------------------------------------------------------------------
extern "C" void kernel_launch(void* const* d_in, const int* in_sizes, int n_in,
                              void* d_out, int out_size, void* d_ws, size_t ws_size,
                              hipStream_t stream) {
    const float* x    = (const float*)d_in[0];
    const float* idd  = (const float*)d_in[1];
    // d_in[2] level, d_in[3] edge_index: tree is a fixed complete binary heap -> unused
    const float* cw   = (const float*)d_in[4];
    const float* cb   = (const float*)d_in[5];
    const float* clw  = (const float*)d_in[6];
    const float* clb  = (const float*)d_in[7];
    const float* Wi   = (const float*)d_in[8];
    const float* bi   = (const float*)d_in[9];
    const float* Wf   = (const float*)d_in[10];
    const float* bff  = (const float*)d_in[11];
    const float* Wu   = (const float*)d_in[12];
    const float* bu   = (const float*)d_in[13];
    const float* Wo   = (const float*)d_in[14];
    const float* bo   = (const float*)d_in[15];
    const float* fc0w = (const float*)d_in[16];
    const float* fc0b = (const float*)d_in[17];
    const float* fc1w = (const float*)d_in[18];
    const float* fc1b = (const float*)d_in[19];
    const float* fc2w = (const float*)d_in[20];
    const float* fc2b = (const float*)d_in[21];
    float* out = (float*)d_out;

    size_t off = 0;
    auto nextp = [&](size_t bytes) -> void* {
        off = (off + 255) & ~(size_t)255;
        void* p = (char*)d_ws + off;
        off += bytes;
        return p;
    };
    __bf16* feat = (__bf16*)nextp((size_t)8192 * FD * 2);        // [8192, 3232] bf16
    float*  cbuf = (float*) nextp((size_t)NNODES * DDIM * 4);    // cell state f32
    __bf16* Wi_b = (__bf16*)nextp((size_t)DDIM * KA * 2);
    __bf16* Wf_b = (__bf16*)nextp((size_t)DDIM * KA * 2);
    __bf16* Wu_b = (__bf16*)nextp((size_t)DDIM * KA * 2);
    __bf16* Wo_b = (__bf16*)nextp((size_t)DDIM * KA * 2);
    __bf16* f0a  = (__bf16*)nextp((size_t)64 * FD * 2);
    __bf16* f0b  = (__bf16*)nextp((size_t)64 * FD * 2);
    __bf16* f0c  = (__bf16*)nextp((size_t)64 * FD * 2);
    __bf16* Aiou = (__bf16*)nextp((size_t)2048 * KA * 2);
    __bf16* Afl  = (__bf16*)nextp((size_t)2048 * KA * 2);
    __bf16* Afr  = (__bf16*)nextp((size_t)2048 * KA * 2);
    float*  Gi   = (float*) nextp((size_t)2048 * DDIM * 4);
    float*  Go   = (float*) nextp((size_t)2048 * DDIM * 4);
    float*  Gu   = (float*) nextp((size_t)2048 * DDIM * 4);
    float*  Gfl  = (float*) nextp((size_t)2048 * DDIM * 4);
    float*  Gfr  = (float*) nextp((size_t)2048 * DDIM * 4);
    float*  y0   = (float*) nextp((size_t)8192 * 64 * 4);
    float*  y1   = (float*) nextp((size_t)8192 * 64 * 4);
    float*  y2   = (float*) nextp((size_t)8192 * 64 * 4);
    (void)ws_size; (void)in_sizes; (void)n_in; (void)out_size;

    auto blks = [](long n) { return (unsigned)((n + 255) / 256); };

    // init + weight conversions
    k_init<<<blks((long)NNODES * DDIM), 256, 0, stream>>>(cbuf, feat + (size_t)8191 * FD);
    k_cvt<<<blks((long)DDIM * KA), 256, 0, stream>>>(Wi, Wi_b, DDIM * KA);
    k_cvt<<<blks((long)DDIM * KA), 256, 0, stream>>>(Wf, Wf_b, DDIM * KA);
    k_cvt<<<blks((long)DDIM * KA), 256, 0, stream>>>(Wu, Wu_b, DDIM * KA);
    k_cvt<<<blks((long)DDIM * KA), 256, 0, stream>>>(Wo, Wo_b, DDIM * KA);
    k_cvt_fc0<<<blks(64L * FD), 256, 0, stream>>>(fc0w, f0a, 0);
    k_cvt_fc0<<<blks(64L * FD), 256, 0, stream>>>(fc0w, f0b, FD);
    k_cvt_fc0<<<blks(64L * FD), 256, 0, stream>>>(fc0w, f0c, 2 * FD);
    k_cvt_idd<<<blks((long)NNODES * 32), 256, 0, stream>>>(idd, feat);

    // conv front-end (x_int for all nodes, leaf h)
    k_conv<<<blks((long)NNODES * 200), 256, 0, stream>>>(x, cw, cb, clw, clb, feat);

    // bottom-up tree-LSTM: level l has M = 2^(12-l) nodes starting at M-1
    for (int lev = 1; lev <= 12; ++lev) {
        int M = 1 << (12 - lev);
        int base = M - 1;
        int Mpad = (M + 255) & ~255;
        if (Mpad < 256) Mpad = 256;
        k_gather<<<blks((long)Mpad * KA), 256, 0, stream>>>(feat, idd, Aiou, Afl, Afr,
                                                            base, M, Mpad);
        dim3 g(DDIM / 64, Mpad / 256);
        k_gemm<<<g, 256, 0, stream>>>(Aiou, Wi_b, Gi, KA, DDIM);
        k_gemm<<<g, 256, 0, stream>>>(Aiou, Wo_b, Go, KA, DDIM);
        k_gemm<<<g, 256, 0, stream>>>(Aiou, Wu_b, Gu, KA, DDIM);
        k_gemm<<<g, 256, 0, stream>>>(Afl,  Wf_b, Gfl, KA, DDIM);
        k_gemm<<<g, 256, 0, stream>>>(Afr,  Wf_b, Gfr, KA, DDIM);
        k_lstm<<<blks((long)M * DDIM), 256, 0, stream>>>(Gi, Go, Gu, Gfl, Gfr,
                                                         bi, bo, bu, bff,
                                                         cbuf, feat, base, M);
    }

    // fc0 decomposed into three GEMMs on feat [8192, 3232]
    {
        dim3 g(1, 8192 / 256);
        k_gemm<<<g, 256, 0, stream>>>(feat, f0a, y0, FD, 64);
        k_gemm<<<g, 256, 0, stream>>>(feat, f0b, y1, FD, 64);
        k_gemm<<<g, 256, 0, stream>>>(feat, f0c, y2, FD, 64);
    }

    // combine + fc1 + fc2
    k_tail<<<NNODES, 64, 0, stream>>>(y0, y1, y2, fc0b, fc1w, fc1b, fc2w, fc2b, out);
}
